// CSPN_9466107920921
// MI455X (gfx1250) — matrix operand order repaired and checked by reading, not compiled, not run on hardware
//
#include <hip/hip_runtime.h>
#include <hip/hip_bf16.h>
#include <math.h>

typedef __attribute__((ext_vector_type(16))) _Float16 v16h;
typedef __attribute__((ext_vector_type(8)))  float    v8f;

#define BATCH 2
#define CHN   32
#define HH    256
#define WW    512
#define HW    (HH*WW)
#define NPIX  (BATCH*HW)
#define TILE_W 128
#define EPSV    1e-6f
#define BN_EPSV 1e-5f

__device__ __forceinline__ int clampi(int v, int lo, int hi) {
    return v < lo ? lo : (v > hi ? hi : v);
}

// ---------------------------------------------------------------------------
// Fold BN (or plain bias) into f16 weights, swizzled directly into the
// per-lane WMMA A-fragment layout:
//   wfrag[ ((mt*9 + kpos)*CB + cbi) * 32 + lane ][ q ]   (16 halves per lane)
// where lane: M = lane%16, g = lane/16; K = 16g+q.
// A fragment load in the conv kernel is then a single contiguous 32B read.
// ---------------------------------------------------------------------------
__global__ void fold_weights_frag(const float* __restrict__ w,
                                  const float* __restrict__ bng, const float* __restrict__ bnb,
                                  const float* __restrict__ bnm, const float* __restrict__ bnv,
                                  const float* __restrict__ bias,
                                  _Float16* __restrict__ wfrag, float* __restrict__ biasf,
                                  int Cout, int Cin, int CoutPad, int CinPad, int hasBN)
{
    int idx = blockIdx.x * blockDim.x + threadIdx.x;
    int CB = CinPad / 32;
    int total = CoutPad * 9 * CinPad;          // == (CoutPad/16)*9*CB*512
    if (idx < total) {
        int q    = idx & 15;
        int lane = (idx >> 4) & 31;
        int r    = idx >> 9;                   // (mt*9 + kpos)*CB + cbi
        int cbi  = r % CB;
        int kr   = r / CB;
        int kpos = kr % 9;
        int mt   = kr / 9;
        int g    = lane >> 4;
        int M    = lane & 15;
        int K    = 16 * g + q;
        int co   = mt * 16 + M;
        int ci   = cbi * 32 + K;
        float val = 0.0f;
        if (co < Cout && ci < Cin) {
            float scale = hasBN ? bng[co] * rsqrtf(bnv[co] + BN_EPSV) : 1.0f;
            val = w[((size_t)co * Cin + ci) * 9 + kpos] * scale;   // (cout,cin,3,3)
        }
        wfrag[idx] = (_Float16)val;
    }
    if (idx < CoutPad) {
        float bv = 0.0f;
        if (idx < Cout) {
            if (hasBN) {
                float scale = bng[idx] * rsqrtf(bnv[idx] + BN_EPSV);
                bv = bnb[idx] - bnm[idx] * scale;
            } else {
                bv = bias[idx];
            }
        }
        biasf[idx] = bv;
    }
}

// ---------------------------------------------------------------------------
// 3x3 conv ('SAME', zero pad) as implicit GEMM on v_wmma_f32_16x16x32_f16.
// Block: 256 threads = 8 waves; each wave owns 16 output pixels of one row.
// LDS tile channel-last: [3 rows][TILE_W+2 cols][CINPAD ch] of f16 so that a
// B fragment (16 consecutive K values for one pixel) is 32 contiguous bytes.
// Compile-time CINPAD / NMT => all acc[] indices constant => no spills.
// ---------------------------------------------------------------------------
template <int CINPAD, int NMT>
__global__ __launch_bounds__(256)
void conv3x3_wmma(const float* __restrict__ in, const _Float16* __restrict__ wfrag,
                  const float* __restrict__ bias, float* __restrict__ out,
                  int Cin, int Cout, int act)
{
    constexpr int CB = CINPAD / 32;
    constexpr int LW = TILE_W + 2;
    __shared__ __align__(32) _Float16 lds[3 * LW * CINPAD];

    const int b  = blockIdx.z;
    const int h  = blockIdx.y;
    const int w0 = blockIdx.x * TILE_W;

    __builtin_prefetch(wfrag, 0, 0);   // global_prefetch of the reused weights

    // Stage fp32 -> f16 into LDS (coalesced global read, transposed LDS write).
    constexpr int total = 3 * CINPAD * LW;
    for (int idx = threadIdx.x; idx < total; idx += 256) {
        int col = idx % LW;           // consecutive threads -> consecutive cols
        int t   = idx / LW;
        int c   = t % CINPAD;
        int r   = t / CINPAD;         // 0..2 (row h-1 .. h+1)
        int gh  = h  + r   - 1;
        int gw  = w0 + col - 1;
        float v = 0.0f;
        if (c < Cin && gh >= 0 && gh < HH && gw >= 0 && gw < WW)
            v = in[((size_t)(b * Cin + c) * HH + gh) * WW + gw];
        lds[(r * LW + col) * CINPAD + c] = (_Float16)v;
    }
    __syncthreads();

    const int lane = threadIdx.x & 31;
    const int wv   = threadIdx.x >> 5;   // wave id: which 16-pixel tile
    const int nl   = lane & 15;
    const int g    = lane >> 4;

    v8f zero = {0.f, 0.f, 0.f, 0.f, 0.f, 0.f, 0.f, 0.f};
    v8f acc[NMT];
    #pragma unroll
    for (int mt = 0; mt < NMT; ++mt) acc[mt] = zero;

    #pragma unroll
    for (int kpos = 0; kpos < 9; ++kpos) {
        const int ky = kpos / 3;
        const int kx = kpos % 3;
        #pragma unroll
        for (int cbi = 0; cbi < CB; ++cbi) {
            // B fragment: pixel = wave tile + lane%16, K = cbi*32 + 16g + q.
            const _Float16* bp =
                lds + ((size_t)(ky * LW + (wv * 16 + nl + kx))) * CINPAD + cbi * 32 + 16 * g;
            v16h bf = *reinterpret_cast<const v16h*>(bp);
            #pragma unroll
            for (int mt = 0; mt < NMT; ++mt) {
                const _Float16* ap =
                    wfrag + (((size_t)(mt * 9 + kpos) * CB + cbi) * 32 + lane) * 16;
                v16h af = *reinterpret_cast<const v16h*>(ap);
                acc[mt] = __builtin_amdgcn_wmma_f32_16x16x32_f16(
                    false, af, false, bf, (short)0, acc[mt], false, false);
            }
        }
    }

    // D layout: M(cout) = vv + 8g, N(pixel) = lane%16.
    const int wpix = w0 + wv * 16 + nl;
    #pragma unroll
    for (int mt = 0; mt < NMT; ++mt) {
        #pragma unroll
        for (int vv = 0; vv < 8; ++vv) {
            int co = mt * 16 + vv + 8 * g;
            if (co < Cout) {
                float val = acc[mt][vv] + bias[co];
                if (act == 1) val = fmaxf(val, 0.0f);
                out[((size_t)(b * Cout + co) * HH + h) * WW + wpix] = val;
            }
        }
    }
}

// ---------------------------------------------------------------------------
// gen_kernel tail: normalize |.|-sum, insert middle weight (1 - sum).
// raw: (B,Km1,H,W)  full: (B,KK,H,W)
// ---------------------------------------------------------------------------
__global__ void genkernel_normalize(const float* __restrict__ raw,
                                    float* __restrict__ full, int Km1, int KK)
{
    int idx = blockIdx.x * blockDim.x + threadIdx.x;
    if (idx >= NPIX) return;
    int b = idx / HW, hw = idx % HW;
    float s = 0.0f;
    for (int c = 0; c < Km1; ++c)
        s += fabsf(raw[((size_t)(b * Km1 + c)) * HW + hw]);
    float inv = 1.0f / (s + EPSV);
    int half = Km1 / 2;
    float ssum = 0.0f;
    for (int c = 0; c < Km1; ++c) {
        float wv = raw[((size_t)(b * Km1 + c)) * HW + hw] * inv;
        ssum += wv;
        int dst = (c < half) ? c : c + 1;
        full[((size_t)(b * KK + dst)) * HW + hw] = wv;
    }
    full[((size_t)(b * KK + half)) * HW + hw] = 1.0f - ssum;
}

// mask = sigmoid(x) * (h0 > 0.001)
__global__ void mask_sigmoid_valid(const float* __restrict__ x,
                                   const float* __restrict__ h0,
                                   float* __restrict__ outm)
{
    int idx = blockIdx.x * blockDim.x + threadIdx.x;
    if (idx >= 3 * NPIX) return;
    int b  = idx / (3 * HW);
    int hw = idx % HW;
    float s = 1.0f / (1.0f + expf(-x[idx]));
    float valid = (h0[(size_t)b * HW + hw] > 0.001f) ? 1.0f : 0.0f;
    outm[idx] = s * valid;
}

// softmax over the 3-channel axis, layout (B,3,H,W)
__global__ void softmax3(const float* __restrict__ x, float* __restrict__ y)
{
    int idx = blockIdx.x * blockDim.x + threadIdx.x;
    if (idx >= NPIX) return;
    int b = idx / HW, hw = idx % HW;
    size_t i0 = ((size_t)(b * 3 + 0)) * HW + hw;
    size_t i1 = ((size_t)(b * 3 + 1)) * HW + hw;
    size_t i2 = ((size_t)(b * 3 + 2)) * HW + hw;
    float x0 = x[i0], x1 = x[i1], x2 = x[i2];
    float m = fmaxf(x0, fmaxf(x1, x2));
    float e0 = expf(x0 - m), e1 = expf(x1 - m), e2 = expf(x2 - m);
    float inv = 1.0f / (e0 + e1 + e2);
    y[i0] = e0 * inv; y[i1] = e1 * inv; y[i2] = e2 * inv;
}

// hnA[:,0..2] = hn ; hns[:,0] = hn
__global__ void init_hn(const float* __restrict__ hn,
                        float* __restrict__ hnA, float* __restrict__ hns)
{
    int idx = blockIdx.x * blockDim.x + threadIdx.x;
    if (idx >= NPIX) return;
    int b = idx / HW, hw = idx % HW;
    float v = hn[idx];
    hnA[((size_t)(b * 3 + 0)) * HW + hw] = v;
    hnA[((size_t)(b * 3 + 1)) * HW + hw] = v;
    hnA[((size_t)(b * 3 + 2)) * HW + hw] = v;
    hns[((size_t)(b * 3 + 0)) * HW + hw] = v;
}

// One CSPN step for all three branches (edge-clamped local conv + mask blend).
__global__ void cspn_iter(const float* __restrict__ hnIn, float* __restrict__ hnOut,
                          const float* __restrict__ w3f, const float* __restrict__ w5f,
                          const float* __restrict__ w7f, const float* __restrict__ mk,
                          const float* __restrict__ h0)
{
    int idx = blockIdx.x * blockDim.x + threadIdx.x;
    if (idx >= NPIX) return;
    int b = idx / HW, hw = idx % HW;
    int h = hw / WW, w = hw % WW;
    float h0v = h0[idx];

    // K = 3
    {
        const float* hp = hnIn + ((size_t)(b * 3 + 0)) * HW;
        float a = 0.0f;
        #pragma unroll
        for (int ky = 0; ky < 3; ++ky) {
            int sh = clampi(h + ky - 1, 0, HH - 1);
            #pragma unroll
            for (int kx = 0; kx < 3; ++kx) {
                int sw = clampi(w + kx - 1, 0, WW - 1);
                a += hp[(size_t)sh * WW + sw] *
                     w3f[((size_t)(b * 9 + ky * 3 + kx)) * HW + hw];
            }
        }
        float m = mk[((size_t)(b * 3 + 0)) * HW + hw];
        hnOut[((size_t)(b * 3 + 0)) * HW + hw] = (1.0f - m) * a + m * h0v;
    }
    // K = 5
    {
        const float* hp = hnIn + ((size_t)(b * 3 + 1)) * HW;
        float a = 0.0f;
        #pragma unroll
        for (int ky = 0; ky < 5; ++ky) {
            int sh = clampi(h + ky - 2, 0, HH - 1);
            #pragma unroll
            for (int kx = 0; kx < 5; ++kx) {
                int sw = clampi(w + kx - 2, 0, WW - 1);
                a += hp[(size_t)sh * WW + sw] *
                     w5f[((size_t)(b * 25 + ky * 5 + kx)) * HW + hw];
            }
        }
        float m = mk[((size_t)(b * 3 + 1)) * HW + hw];
        hnOut[((size_t)(b * 3 + 1)) * HW + hw] = (1.0f - m) * a + m * h0v;
    }
    // K = 7
    {
        const float* hp = hnIn + ((size_t)(b * 3 + 2)) * HW;
        float a = 0.0f;
        #pragma unroll
        for (int ky = 0; ky < 7; ++ky) {
            int sh = clampi(h + ky - 3, 0, HH - 1);
            #pragma unroll
            for (int kx = 0; kx < 7; ++kx) {
                int sw = clampi(w + kx - 3, 0, WW - 1);
                a += hp[(size_t)sh * WW + sw] *
                     w7f[((size_t)(b * 49 + ky * 7 + kx)) * HW + hw];
            }
        }
        float m = mk[((size_t)(b * 3 + 2)) * HW + hw];
        hnOut[((size_t)(b * 3 + 2)) * HW + hw] = (1.0f - m) * a + m * h0v;
    }
}

// hns[:,ch] = c3*hn3 + c5*hn5 + c7*hn7
__global__ void combine_ch(const float* __restrict__ hnbuf,
                           const float* __restrict__ ckb,
                           float* __restrict__ hns, int ch)
{
    int idx = blockIdx.x * blockDim.x + threadIdx.x;
    if (idx >= NPIX) return;
    int b = idx / HW, hw = idx % HW;
    float v = 0.0f;
    #pragma unroll
    for (int c = 0; c < 3; ++c)
        v += ckb[((size_t)(b * 3 + c)) * HW + hw] *
             hnbuf[((size_t)(b * 3 + c)) * HW + hw];
    hns[((size_t)(b * 3 + ch)) * HW + hw] = v;
}

// cat = concat(fout[32ch], hns[3ch]) -> (B,35,H,W)
__global__ void build_cat(const float* __restrict__ fout,
                          const float* __restrict__ hns, float* __restrict__ cat)
{
    long long idx = (long long)blockIdx.x * blockDim.x + threadIdx.x;
    if (idx >= (long long)35 * NPIX) return;
    int hw = (int)(idx % HW);
    int r  = (int)(idx / HW);
    int c  = r % 35;
    int b  = r / 35;
    float v = (c < 32) ? fout[((size_t)(b * 32 + c)) * HW + hw]
                       : hns[((size_t)(b * 3 + (c - 32))) * HW + hw];
    cat[idx] = v;
}

// out = sum_c wt[c] * hns[c]
__global__ void final_out(const float* __restrict__ wt,
                          const float* __restrict__ hns, float* __restrict__ out)
{
    int idx = blockIdx.x * blockDim.x + threadIdx.x;
    if (idx >= NPIX) return;
    int b = idx / HW, hw = idx % HW;
    float v = 0.0f;
    #pragma unroll
    for (int c = 0; c < 3; ++c)
        v += wt[((size_t)(b * 3 + c)) * HW + hw] *
             hns[((size_t)(b * 3 + c)) * HW + hw];
    out[idx] = v;
}

// ---------------------------------------------------------------------------
// Host orchestration
// ---------------------------------------------------------------------------
extern "C" void kernel_launch(void* const* d_in, const int* in_sizes, int n_in,
                              void* d_out, int out_size, void* d_ws, size_t ws_size,
                              hipStream_t stream)
{
    (void)in_sizes; (void)n_in; (void)out_size; (void)ws_size;

    const float* fout = (const float*)d_in[0];
    const float* hn   = (const float*)d_in[1];
    const float* h0   = (const float*)d_in[2];
    auto P = [&](int i) { return (const float*)d_in[i]; };
    // pytree order (nested dicts key-sorted):
    // ck: 3=bn.b 4=bn.g 5=bn.m 6=bn.v 7=w   8=c2.b 9=c2.w
    // ct: 10..14 (bn.b,g,m,v,w) 15=c2.b 16=c2.w
    // mask: 17..21, 22=c2.b 23=c2.w
    // w3: 24..28 c1, 29..33 c2(bn+w) ; w5: 34..43 ; w7: 44..53

    float* ws = (float*)d_ws;
    size_t off = 0;
    auto A = [&](size_t n) { float* p = ws + off; off += n; return p; };
    float* t1  = A((size_t)32 * NPIX);
    float* raw = A((size_t)48 * NPIX);   // conv2 scratch; later reused as `cat`
    float* w3k = A((size_t)9  * NPIX);
    float* w5k = A((size_t)25 * NPIX);
    float* w7k = A((size_t)49 * NPIX);
    float* mkb = A((size_t)3  * NPIX);
    float* ckb = A((size_t)3  * NPIX);
    float* hnA = A((size_t)3  * NPIX);
    float* hnB = A((size_t)3  * NPIX);
    float* hns = A((size_t)3  * NPIX);
    float* wt  = A((size_t)3  * NPIX);

    struct WB { _Float16* w; float* b; int cout, cin, coutPad, cinPad; };
    auto AW = [&](int cout, int cin) -> WB {
        WB r;
        r.cout = cout; r.cin = cin;
        r.coutPad = ((cout + 15) / 16) * 16;
        r.cinPad  = ((cin  + 31) / 32) * 32;
        r.w = (_Float16*)(ws + off);
        off += ((size_t)r.coutPad * 9 * r.cinPad + 1) / 2;  // halves -> float slots
        r.b = ws + off; off += r.coutPad;
        return r;
    };
    WB w3c1 = AW(32, 32), w3c2 = AW(8, 32);
    WB w5c1 = AW(32, 32), w5c2 = AW(24, 32);
    WB w7c1 = AW(32, 32), w7c2 = AW(48, 32);
    WB mkc1 = AW(32, 32), mkc2 = AW(3, 32);
    WB ckc1 = AW(32, 32), ckc2 = AW(3, 32);
    WB ctc1 = AW(32, 35), ctc2 = AW(3, 32);

    auto fold = [&](WB& wb, const float* w, const float* bg, const float* bb,
                    const float* bm, const float* bv, const float* bias, int hasBN) {
        int total = wb.coutPad * 9 * wb.cinPad;
        fold_weights_frag<<<(total + 255) / 256, 256, 0, stream>>>(
            w, bg, bb, bm, bv, bias, wb.w, wb.b,
            wb.cout, wb.cin, wb.coutPad, wb.cinPad, hasBN);
    };
    fold(w3c1, P(28), P(25), P(24), P(26), P(27), nullptr, 1);
    fold(w3c2, P(33), P(30), P(29), P(31), P(32), nullptr, 1);
    fold(w5c1, P(38), P(35), P(34), P(36), P(37), nullptr, 1);
    fold(w5c2, P(43), P(40), P(39), P(41), P(42), nullptr, 1);
    fold(w7c1, P(48), P(45), P(44), P(46), P(47), nullptr, 1);
    fold(w7c2, P(53), P(50), P(49), P(51), P(52), nullptr, 1);
    fold(mkc1, P(21), P(18), P(17), P(19), P(20), nullptr, 1);
    fold(mkc2, P(23), nullptr, nullptr, nullptr, nullptr, P(22), 0);
    fold(ckc1, P(7),  P(4),  P(3),  P(5),  P(6),  nullptr, 1);
    fold(ckc2, P(9),  nullptr, nullptr, nullptr, nullptr, P(8), 0);
    fold(ctc1, P(14), P(11), P(10), P(12), P(13), nullptr, 1);
    fold(ctc2, P(16), nullptr, nullptr, nullptr, nullptr, P(15), 0);

    auto conv = [&](const float* in, WB& wb, float* out_, int act) {
        dim3 grid(WW / TILE_W, HH, BATCH);
        if (wb.cinPad == 32 && wb.coutPad == 16)
            conv3x3_wmma<32, 1><<<grid, 256, 0, stream>>>(
                in, wb.w, wb.b, out_, wb.cin, wb.cout, act);
        else if (wb.cinPad == 32 && wb.coutPad == 32)
            conv3x3_wmma<32, 2><<<grid, 256, 0, stream>>>(
                in, wb.w, wb.b, out_, wb.cin, wb.cout, act);
        else if (wb.cinPad == 32 && wb.coutPad == 48)
            conv3x3_wmma<32, 3><<<grid, 256, 0, stream>>>(
                in, wb.w, wb.b, out_, wb.cin, wb.cout, act);
        else
            conv3x3_wmma<64, 2><<<grid, 256, 0, stream>>>(
                in, wb.w, wb.b, out_, wb.cin, wb.cout, act);
    };
    auto B1 = [](long long n) { return (unsigned)((n + 255) / 256); };

    // affinity kernels w3/w5/w7
    conv(fout, w3c1, t1, 1);
    conv(t1, w3c2, raw, 1);
    genkernel_normalize<<<B1(NPIX), 256, 0, stream>>>(raw, w3k, 8, 9);
    conv(fout, w5c1, t1, 1);
    conv(t1, w5c2, raw, 1);
    genkernel_normalize<<<B1(NPIX), 256, 0, stream>>>(raw, w5k, 24, 25);
    conv(fout, w7c1, t1, 1);
    conv(t1, w7c2, raw, 1);
    genkernel_normalize<<<B1(NPIX), 256, 0, stream>>>(raw, w7k, 48, 49);

    // mask
    conv(fout, mkc1, t1, 1);
    conv(t1, mkc2, raw, 0);
    mask_sigmoid_valid<<<B1(3LL * NPIX), 256, 0, stream>>>(raw, h0, mkb);

    // ck
    conv(fout, ckc1, t1, 1);
    conv(t1, ckc2, raw, 0);
    softmax3<<<B1(NPIX), 256, 0, stream>>>(raw, ckb);

    // CSPN iterations
    init_hn<<<B1(NPIX), 256, 0, stream>>>(hn, hnA, hns);
    float* cur = hnA;
    float* nxt = hnB;
    for (int i = 0; i < 12; ++i) {
        cspn_iter<<<B1(NPIX), 256, 0, stream>>>(cur, nxt, w3k, w5k, w7k, mkb, h0);
        float* t = cur; cur = nxt; nxt = t;
        if (i == 5)
            combine_ch<<<B1(NPIX), 256, 0, stream>>>(cur, ckb, hns, 1);
    }
    combine_ch<<<B1(NPIX), 256, 0, stream>>>(cur, ckb, hns, 2);

    // final temporal attention
    float* cat = raw;  // 48-channel scratch reused for 35-channel concat input
    build_cat<<<B1(35LL * NPIX), 256, 0, stream>>>(fout, hns, cat);
    conv(cat, ctc1, t1, 1);
    conv(t1, ctc2, mkb, 0);  // mkb reused as 3-channel logits buffer
    softmax3<<<B1(NPIX), 256, 0, stream>>>(mkb, wt);
    final_out<<<B1(NPIX), 256, 0, stream>>>(wt, hns, (float*)d_out);
}
